// MiniDecoderBlock_61117384622565
// MI455X (gfx1250) — compile-verified, matrix operand order checked by reading, not run on hardware
//
#include <hip/hip_runtime.h>
#include <hip/hip_bf16.h>

#define DM   1024
#define NH   16
#define HD   64
#define DI   2048
#define BB   2
#define TT   2048
#define MM   (BB*TT)   // 4096 rows

typedef __bf16 bf16;
typedef __attribute__((ext_vector_type(16))) bf16  v16bf;
typedef __attribute__((ext_vector_type(8)))  bf16  v8bf;
typedef __attribute__((ext_vector_type(8)))  float v8f;

#define AS1 __attribute__((address_space(1)))
typedef const AS1 bf16* gcbf;
typedef const AS1 v8bf* gcv8;

union FragB { v16bf v; v8bf h[2]; };

__device__ inline gcbf to_global(const bf16* p) {
    return (gcbf)(unsigned long long)(uintptr_t)p;
}

// Load a 16x32 bf16 WMMA fragment (A layout; B uses the same layout on W rows,
// since both GEMMs are x @ W^T). For lane L: m/n = L&15, hl = L>>4.
// elements 0..7  = p[0..7]   (K = k0 + 8*hl .. +7)
// elements 8..15 = p[16..23] (K = k0 + 16 + 8*hl .. +7)
__device__ inline v16bf load_frag_g(gcbf p) {
    FragB f;
    f.h[0] = ((gcv8)p)[0];
    f.h[1] = ((gcv8)p)[2];
    return f.v;
}

__device__ inline v8f wmma_bf16(v16bf a, v16bf b, v8f c) {
    return __builtin_amdgcn_wmma_f32_16x16x32_bf16(false, a, false, b, (short)0, c, false, false);
}

// ---------------- generic GEMM: acc[m][n] = sum_k A[m][k]*W[n][k] -------------
// one wave computes a 64x64 tile; grid = (N/64, M/64), block = 32
// epilogue modes (compile-time):
//   0: C[idx] = acc (+ R[idx] if HASR)                    (fp32 out)
//   1: ACT[idx] = bf16( silu(G[idx]) * acc )              (fused up*silu(gate))
//   2: qkv split: row->(b,t), col->(which,h,d); writes q (scaled 1/8), k, vT
template<int MODE, bool HASR>
__global__ __launch_bounds__(32)
void gemm_bf16_wmma(const bf16* __restrict__ A, const bf16* __restrict__ W,
                    const float* __restrict__ R, float* __restrict__ C,
                    const float* __restrict__ G, bf16* __restrict__ ACT,
                    bf16* __restrict__ q, bf16* __restrict__ k,
                    bf16* __restrict__ vT,
                    int M, int N, int K) {
    const int lane = threadIdx.x;
    const int hl = lane >> 4, ln = lane & 15;
    const int m0 = blockIdx.y * 64, n0 = blockIdx.x * 64;

    // running fragment pointers in explicit global address space
    gcbf pa0 = to_global(A + (size_t)(m0 + 0 * 16 + ln) * K + 8 * hl);
    gcbf pa1 = to_global(A + (size_t)(m0 + 1 * 16 + ln) * K + 8 * hl);
    gcbf pa2 = to_global(A + (size_t)(m0 + 2 * 16 + ln) * K + 8 * hl);
    gcbf pa3 = to_global(A + (size_t)(m0 + 3 * 16 + ln) * K + 8 * hl);
    gcbf pb0 = to_global(W + (size_t)(n0 + 0 * 16 + ln) * K + 8 * hl);
    gcbf pb1 = to_global(W + (size_t)(n0 + 1 * 16 + ln) * K + 8 * hl);
    gcbf pb2 = to_global(W + (size_t)(n0 + 2 * 16 + ln) * K + 8 * hl);
    gcbf pb3 = to_global(W + (size_t)(n0 + 3 * 16 + ln) * K + 8 * hl);

    v8f acc[4][4] = {};
    for (int k0 = 0; k0 < K; k0 += 32) {
        v16bf a[4], b[4];
        a[0] = load_frag_g(pa0); pa0 += 32;
        a[1] = load_frag_g(pa1); pa1 += 32;
        a[2] = load_frag_g(pa2); pa2 += 32;
        a[3] = load_frag_g(pa3); pa3 += 32;
        b[0] = load_frag_g(pb0); pb0 += 32;
        b[1] = load_frag_g(pb1); pb1 += 32;
        b[2] = load_frag_g(pb2); pb2 += 32;
        b[3] = load_frag_g(pb3); pb3 += 32;
#pragma unroll
        for (int i = 0; i < 4; i++)
#pragma unroll
            for (int j = 0; j < 4; j++)
                acc[i][j] = wmma_bf16(a[i], b[j], acc[i][j]);
    }

#pragma unroll
    for (int i = 0; i < 4; i++)
#pragma unroll
        for (int j = 0; j < 4; j++)
#pragma unroll
            for (int r = 0; r < 8; r++) {
                const int row = m0 + i * 16 + r + 8 * hl;
                const int col = n0 + j * 16 + ln;
                const size_t idx = (size_t)row * N + col;
                const float v = acc[i][j][r];
                if (MODE == 0) {
                    float o = v;
                    if (HASR) o += R[idx];
                    C[idx] = o;
                } else if (MODE == 1) {
                    float gg = G[idx];
                    float s = gg / (1.0f + __expf(-gg));
                    ACT[idx] = (bf16)(s * v);
                } else {
                    const int b_ = row >> 11;          // row / TT
                    const int t  = row & (TT - 1);
                    const int which = col >> 10;
                    const int wi = col & (DM - 1);
                    const int h = wi >> 6, d = wi & (HD - 1);
                    const size_t bh = (size_t)(b_ * NH + h);
                    if (which == 0)      q[(bh * TT + t) * HD + d] = (bf16)(v * 0.125f);
                    else if (which == 1) k[(bh * TT + t) * HD + d] = (bf16)v;
                    else                 vT[(bh * HD + d) * TT + t] = (bf16)v;
                }
            }
}

// ---------------- RMSNorm (fp32 in, bf16 out) --------------------------------
__global__ __launch_bounds__(256)
void rmsnorm_bf16(const float* __restrict__ X, const float* __restrict__ W,
                  bf16* __restrict__ O) {
    int row = blockIdx.x;
    const float* xr = X + (size_t)row * DM;
    float s = 0.f;
    for (int i = threadIdx.x; i < DM; i += 256) { float v = xr[i]; s += v * v; }
    __shared__ float red[8];
    for (int off = 16; off >= 1; off >>= 1) s += __shfl_xor(s, off, 32);
    if ((threadIdx.x & 31) == 0) red[threadIdx.x >> 5] = s;
    __syncthreads();
    if (threadIdx.x == 0) {
        float t = 0.f;
        for (int w = 0; w < 8; w++) t += red[w];
        red[0] = t;
    }
    __syncthreads();
    float scale = rsqrtf(red[0] * (1.0f / DM) + 1e-6f);
    bf16* orow = O + (size_t)row * DM;
    for (int i = threadIdx.x; i < DM; i += 256)
        orow[i] = (bf16)(xr[i] * scale * W[i]);
}

// ---------------- weight fp32 -> bf16 ----------------------------------------
__global__ void f32_to_bf16(const float* __restrict__ S, bf16* __restrict__ D, int n) {
    int i = blockIdx.x * 256 + threadIdx.x;
    if (i < n) D[i] = (bf16)S[i];
}

// ---------------- flash attention: 1 wave per (b,h, 16-query tile) -----------
__global__ __launch_bounds__(32)
void attn_fa(const bf16* __restrict__ Qb, const bf16* __restrict__ Kb,
             const bf16* __restrict__ VTb, bf16* __restrict__ Ob) {
    __shared__ bf16 Pld[16][32];  // P tile staging for C->A layout transpose
    const int lane = threadIdx.x, hl = lane >> 4, ln = lane & 15;
    const int nqt = TT / 16;
    const int bh = blockIdx.x / nqt;
    const int q0 = (blockIdx.x % nqt) * 16;
    const bf16* Q  = Qb  + (size_t)bh * TT * HD;
    const bf16* Kk = Kb  + (size_t)bh * TT * HD;
    const bf16* VT = VTb + (size_t)bh * HD * TT;

    const v16bf aq0 = load_frag_g(to_global(Q + (size_t)(q0 + ln) * HD + 8 * hl));
    const v16bf aq1 = load_frag_g(to_global(Q + (size_t)(q0 + ln) * HD + 32 + 8 * hl));

    v8f o[4] = {};
    float rm[8], rl[8];
#pragma unroll
    for (int r = 0; r < 8; r++) { rm[r] = -1e30f; rl[r] = 0.f; }

    for (int kb = 0; kb < q0 + 16; kb += 32) {
        // S = Q @ K^T for keys [kb, kb+32): two 16x16 C tiles
        v8f s0 = {}, s1 = {};
        {
            v16bf bk = load_frag_g(to_global(Kk + (size_t)(kb + ln) * HD + 8 * hl));
            s0 = wmma_bf16(aq0, bk, s0);
            bk = load_frag_g(to_global(Kk + (size_t)(kb + ln) * HD + 32 + 8 * hl));
            s0 = wmma_bf16(aq1, bk, s0);
            bk = load_frag_g(to_global(Kk + (size_t)(kb + 16 + ln) * HD + 8 * hl));
            s1 = wmma_bf16(aq0, bk, s1);
            bk = load_frag_g(to_global(Kk + (size_t)(kb + 16 + ln) * HD + 32 + 8 * hl));
            s1 = wmma_bf16(aq1, bk, s1);
        }
        // online softmax; C-layout row (q0+r+8*hl) spans the 16 lanes of this half
#pragma unroll
        for (int r = 0; r < 8; r++) {
            int row = q0 + r + 8 * hl;
            float v0 = (kb + ln      <= row) ? s0[r] : -1e30f;   // causal mask
            float v1 = (kb + 16 + ln <= row) ? s1[r] : -1e30f;
            float m = fmaxf(v0, v1);
            m = fmaxf(m, __shfl_xor(m, 1, 16));
            m = fmaxf(m, __shfl_xor(m, 2, 16));
            m = fmaxf(m, __shfl_xor(m, 4, 16));
            m = fmaxf(m, __shfl_xor(m, 8, 16));
            float nm = fmaxf(rm[r], m);
            float p0 = __expf(v0 - nm), p1 = __expf(v1 - nm);
            float alpha = __expf(rm[r] - nm);
            rm[r] = nm;
            float ps = p0 + p1;
            ps += __shfl_xor(ps, 1, 16);
            ps += __shfl_xor(ps, 2, 16);
            ps += __shfl_xor(ps, 4, 16);
            ps += __shfl_xor(ps, 8, 16);
            rl[r] = rl[r] * alpha + ps;
#pragma unroll
            for (int dt = 0; dt < 4; dt++) o[dt][r] *= alpha;
            Pld[r + 8 * hl][ln]      = (bf16)p0;
            Pld[r + 8 * hl][16 + ln] = (bf16)p1;
        }
        __syncthreads();                       // single-wave WG: S_NOP + fence
        FragB ap;
        ap.h[0] = *(const v8bf*)&Pld[ln][8 * hl];
        ap.h[1] = *(const v8bf*)&Pld[ln][16 + 8 * hl];
        __syncthreads();
        // O += P @ V ; V fragments from pre-transposed V^T [d][t]
#pragma unroll
        for (int dt = 0; dt < 4; dt++) {
            v16bf bv = load_frag_g(to_global(VT + (size_t)(dt * 16 + ln) * TT + kb + 8 * hl));
            o[dt] = wmma_bf16(ap.v, bv, o[dt]);
        }
    }
    const int b = bh / NH, h = bh % NH;
#pragma unroll
    for (int dt = 0; dt < 4; dt++)
#pragma unroll
        for (int r = 0; r < 8; r++) {
            int t = q0 + r + 8 * hl;
            float v = o[dt][r] / rl[r];
            Ob[((size_t)(b * TT + t)) * DM + h * HD + dt * 16 + ln] = (bf16)v;
        }
}

// ---------------- host orchestration -----------------------------------------
extern "C" void kernel_launch(void* const* d_in, const int* in_sizes, int n_in,
                              void* d_out, int out_size, void* d_ws, size_t ws_size,
                              hipStream_t stream) {
    const float* x      = (const float*)d_in[0];
    const float* ln1_w  = (const float*)d_in[1];
    const float* ln2_w  = (const float*)d_in[2];
    const float* qkv_w  = (const float*)d_in[3];
    const float* o_w    = (const float*)d_in[4];
    const float* gate_w = (const float*)d_in[5];
    const float* up_w   = (const float*)d_in[6];
    const float* down_w = (const float*)d_in[7];
    float* out = (float*)d_out;
    char* ws = (char*)d_ws;

    // workspace layout (phase-based reuse, ~92 MB peak)
    size_t o_wqkv = 0;                                  // 3*DM*DM bf16
    size_t o_wo   = o_wqkv + (size_t)3 * DM * DM * 2;   // DM*DM bf16
    size_t o_wg   = o_wo   + (size_t)DM * DM * 2;       // DI*DM bf16
    size_t o_wu   = o_wg   + (size_t)DI * DM * 2;       // DI*DM bf16
    size_t o_wd   = o_wu   + (size_t)DI * DM * 2;       // DM*DI bf16
    size_t o_xn   = o_wd   + (size_t)DM * DI * 2;       // MM*DM bf16 (xn1 then xn2)
    size_t o_h1   = o_xn   + (size_t)MM * DM * 2;       // MM*DM f32
    size_t o_big  = o_h1   + (size_t)MM * DM * 4;
    // phase A (attention): q, k, vT (bf16, 8 MB each) + attn (bf16, 8 MB)
    size_t o_q    = o_big;
    size_t o_k    = o_q  + (size_t)MM * DM * 2;
    size_t o_vT   = o_k  + (size_t)MM * DM * 2;
    size_t o_attn = o_vT + (size_t)MM * DM * 2;
    // phase B (FFN): g fp32 (32 MB, reuses q/k/vT/attn), act bf16 (16 MB)
    size_t o_g    = o_big;
    size_t o_act  = o_big + (size_t)MM * DI * 4;

    bf16* wqkv = (bf16*)(ws + o_wqkv);
    bf16* wo   = (bf16*)(ws + o_wo);
    bf16* wg   = (bf16*)(ws + o_wg);
    bf16* wu   = (bf16*)(ws + o_wu);
    bf16* wd   = (bf16*)(ws + o_wd);
    bf16* xn   = (bf16*)(ws + o_xn);
    float* h1  = (float*)(ws + o_h1);
    bf16* qb   = (bf16*)(ws + o_q);
    bf16* kb   = (bf16*)(ws + o_k);
    bf16* vT   = (bf16*)(ws + o_vT);
    bf16* attn = (bf16*)(ws + o_attn);
    float* g   = (float*)(ws + o_g);
    bf16* act  = (bf16*)(ws + o_act);

    // 1. weights fp32 -> bf16
    int nqkvw = 3 * DM * DM, now = DM * DM, ngw = DI * DM;
    f32_to_bf16<<<(nqkvw + 255) / 256, 256, 0, stream>>>(qkv_w, wqkv, nqkvw);
    f32_to_bf16<<<(now + 255) / 256, 256, 0, stream>>>(o_w, wo, now);
    f32_to_bf16<<<(ngw + 255) / 256, 256, 0, stream>>>(gate_w, wg, ngw);
    f32_to_bf16<<<(ngw + 255) / 256, 256, 0, stream>>>(up_w, wu, ngw);
    f32_to_bf16<<<(ngw + 255) / 256, 256, 0, stream>>>(down_w, wd, ngw);

    // 2. xn1 = rmsnorm(x, ln1_w)
    rmsnorm_bf16<<<MM, 256, 0, stream>>>(x, ln1_w, xn);

    // 3. qkv GEMM with fused split epilogue -> q (scaled), k, vT (bf16)
    gemm_bf16_wmma<2, false><<<dim3(3 * DM / 64, MM / 64), 32, 0, stream>>>(
        xn, wqkv, nullptr, nullptr, nullptr, nullptr, qb, kb, vT,
        MM, 3 * DM, DM);

    // 4. flash attention
    attn_fa<<<BB * NH * (TT / 16), 32, 0, stream>>>(qb, kb, vT, attn);

    // 5. h1 = x + attn @ o_w^T
    gemm_bf16_wmma<0, true><<<dim3(DM / 64, MM / 64), 32, 0, stream>>>(
        attn, wo, x, h1, nullptr, nullptr, nullptr, nullptr, nullptr,
        MM, DM, DM);

    // 6. xn2 = rmsnorm(h1, ln2_w)
    rmsnorm_bf16<<<MM, 256, 0, stream>>>(h1, ln2_w, xn);

    // 7. g = xn2 @ gate_w^T (fp32)
    gemm_bf16_wmma<0, false><<<dim3(DI / 64, MM / 64), 32, 0, stream>>>(
        xn, wg, nullptr, g, nullptr, nullptr, nullptr, nullptr, nullptr,
        MM, DI, DM);

    // 8. up GEMM with fused epilogue: act = bf16( silu(g) * (xn2 @ up_w^T) )
    gemm_bf16_wmma<1, false><<<dim3(DI / 64, MM / 64), 32, 0, stream>>>(
        xn, wu, nullptr, nullptr, g, act, nullptr, nullptr, nullptr,
        MM, DI, DM);

    // 9. out = h1 + act @ down_w^T
    gemm_bf16_wmma<0, true><<<dim3(DM / 64, MM / 64), 32, 0, stream>>>(
        act, wd, h1, out, nullptr, nullptr, nullptr, nullptr, nullptr,
        MM, DM, DI);

    (void)in_sizes; (void)n_in; (void)out_size; (void)ws_size;
}
